// QKVAttentionLegacy_51410758533174
// MI455X (gfx1250) — compile-verified, matrix operand order checked by reading, not run on hardware
//
#include <hip/hip_runtime.h>

// ---------------------------------------------------------------------------
// QKV attention for MI455X (gfx1250): flash-attention with f16 WMMA.
//   qkv: (8, 1536, 2048) f32 -> 64 head-batches b, ch=64, L=2048
//   out: (8, 512, 2048) f32  == [b][c][t] with b in 0..63
// Compute-bound (~512 FLOP/byte) -> f16 WMMA; softmax in exp2 domain;
// row-sums computed on the WMMA pipe (P x ones) to unload the VALU.
// ---------------------------------------------------------------------------

typedef __attribute__((ext_vector_type(16))) _Float16 v16h;
typedef __attribute__((ext_vector_type(8)))  _Float16 v8h;
typedef __attribute__((ext_vector_type(8)))  float    v8f;
typedef __attribute__((__vector_size__(16))) int      vsi4;  // async-lds ptr elt

#define SEQ    2048
#define CH     64
#define NHEADB 64
#define SCHUNK 128     // k/v columns per LDS chunk (amortizes softmax fixed cost)
#define TBLOCK 128
#define NWAVES 8
#define OSTRIDE 132    // padded f32 row stride for epilogue transpose

#if defined(__has_builtin)
#if __has_builtin(__builtin_amdgcn_global_load_async_to_lds_b128)
#define USE_ASYNC_V 1
#endif
#endif

// Build a v16h WMMA operand from two contiguous 16-byte LDS chunks.
// Per ISA 16-bit A/B layout: a lane's 16 halves are K=[k0..k0+7] then
// [k0+16..k0+23], where k0 = 32*kstep + 8*(lane>=16). Both 16B aligned.
static __device__ __forceinline__ v16h load_frag(const _Float16* p0,
                                                 const _Float16* p1) {
    union { v16h v; v8h h[2]; } u;
    u.h[0] = *(const v8h*)p0;
    u.h[1] = *(const v8h*)p1;
    return u.v;
}

// XOR-shuffle within each 16-lane half via v_permlane16_b32 (VALU, no LDS).
static __device__ __forceinline__ float xor16(float v, unsigned s0, unsigned s1) {
    return __builtin_bit_cast(float,
        __builtin_amdgcn_permlane16(0u, __builtin_bit_cast(unsigned, v),
                                    s0, s1, false, false));
}
static __device__ __forceinline__ float redmax16(float v) {
    v = fmaxf(v, xor16(v, 0x67452301u, 0xEFCDAB89u));   // ^1
    v = fmaxf(v, xor16(v, 0x54761302u, 0xDCFE98BAu));   // ^2
    v = fmaxf(v, xor16(v, 0x32107654u, 0xBA98FEDCu));   // ^4
    v = fmaxf(v, xor16(v, 0xFEDCBA98u, 0x76543210u));   // ^8
    return v;
}

// ---------------------------------------------------------------------------
// Pass 1: f32 -> f16; fold softmax scale 1/8 AND log2(e) into Q rows so the
// attention kernel can use native v_exp_f32 (exp2) directly.
// ---------------------------------------------------------------------------
__global__ __launch_bounds__(256)
void qkv_cvt_f16(const float* __restrict__ src, _Float16* __restrict__ dst) {
    long i = (long)blockIdx.x * 256 + threadIdx.x;     // float4 index
    float4 f = ((const float4*)src)[i];
    long e = i << 2;
    int r = (int)((e >> 11) % 192);                    // row within head slab
    float s = (r < 64) ? 0.125f * 1.4426950408889634f : 1.0f;
    union { _Float16 h[4]; uint2 u; } p;
    p.h[0] = (_Float16)(f.x * s);
    p.h[1] = (_Float16)(f.y * s);
    p.h[2] = (_Float16)(f.z * s);
    p.h[3] = (_Float16)(f.w * s);
    ((uint2*)dst)[i] = p.u;
}

// ---------------------------------------------------------------------------
// Pass 2: flash attention. grid = (64 heads, 16 q-blocks), 256 threads.
// ---------------------------------------------------------------------------
__global__ __launch_bounds__(256)
void flash_attn_f16(const _Float16* __restrict__ qkvh, float* __restrict__ out) {
    // 80 KB LDS carve (WGP has 320 KB); front 33 KB reused by the epilogue
    // as a padded f32 transpose buffer for coalesced output stores.
    __shared__ __align__(16) unsigned char smem[81920];
    _Float16* qT  = (_Float16*)smem;               // [t][c]   128x64  16 KB
    _Float16* kT  = (_Float16*)(smem + 16384);     // [s][c]   128x64  16 KB
    _Float16* vS  = (_Float16*)(smem + 32768);     // [c][s]   64x128  16 KB
    _Float16* pls = (_Float16*)(smem + 49152);     // per-wave P 16x128 32 KB
    float*    obuf = (float*)smem;                 // epilogue [c][t+pad]

    const int b    = blockIdx.x;
    const int t0   = blockIdx.y * TBLOCK;
    const int tid  = threadIdx.x;
    const int wid  = tid >> 5;
    const int lane = tid & 31;
    const int lp   = lane & 15;
    const int half = lane >> 4;

    const _Float16* qg = qkvh + (long)b * 192 * SEQ;
    const _Float16* kg = qg + (long)64  * SEQ;
    const _Float16* vg = qg + (long)128 * SEQ;

    // ---- stage Q transposed: global [c][t] -> qT[t][c], packed b32 stores.
    for (int g = tid; g < 512; g += 256) {
        int c0 = (g >> 4) * 2;
        int tt = (g & 15) * 8;
        uint4 u0 = *(const uint4*)(qg + (long)c0 * SEQ + t0 + tt);
        uint4 u1 = *(const uint4*)(qg + (long)(c0 + 1) * SEQ + t0 + tt);
        const _Float16* h0 = (const _Float16*)&u0;
        const _Float16* h1 = (const _Float16*)&u1;
#pragma unroll
        for (int j = 0; j < 8; ++j) {
            union { _Float16 h[2]; unsigned u; } w;
            w.h[0] = h0[j]; w.h[1] = h1[j];
            *(unsigned*)&qT[(tt + j) * CH + c0] = w.u;
        }
    }
    __syncthreads();

    // Persistent A-fragments of Q^T (M=16 t-rows, K=64 channels -> 2 frags).
    v16h qa[2];
#pragma unroll
    for (int kk = 0; kk < 2; ++kk) {
        const _Float16* base = &qT[(wid * 16 + lp) * CH];
        int k0 = kk * 32 + half * 8;
        qa[kk] = load_frag(base + k0, base + k0 + 16);
    }

    // All-ones B fragment: row-sum of P via the WMMA pipe.
    v16h ones;
#pragma unroll
    for (int i = 0; i < 16; ++i) ones[i] = (_Float16)1.0f;

    const v8f zero8 = {0.f, 0.f, 0.f, 0.f, 0.f, 0.f, 0.f, 0.f};
    v8f  o[4];
    float mrow[8], lrow[8];
#pragma unroll
    for (int j = 0; j < 4; ++j) o[j] = zero8;
#pragma unroll
    for (int r = 0; r < 8; ++r) { mrow[r] = -3.0e38f; lrow[r] = 0.f; }

    _Float16* pw = &pls[wid * 16 * SCHUNK];

    for (int s0 = 0; s0 < SEQ; s0 += SCHUNK) {
        __syncthreads();                             // chunk consumers done

        // ---- stage K^T (packed transpose) and V (async direct copy) ----
        for (int g = tid; g < 512; g += 256) {       // 32 c-pairs x 16 s-grps
            int c0 = (g >> 4) * 2;
            int ss = (g & 15) * 8;
            uint4 u0 = *(const uint4*)(kg + (long)c0 * SEQ + s0 + ss);
            uint4 u1 = *(const uint4*)(kg + (long)(c0 + 1) * SEQ + s0 + ss);
            const _Float16* h0 = (const _Float16*)&u0;
            const _Float16* h1 = (const _Float16*)&u1;
#pragma unroll
            for (int j = 0; j < 8; ++j) {
                union { _Float16 h[2]; unsigned u; } w;
                w.h[0] = h0[j]; w.h[1] = h1[j];
                *(unsigned*)&kT[(ss + j) * CH + c0] = w.u;
            }
        }
        for (int g = tid; g < 1024; g += 256) {      // 64 rows x 16 s-grps
            int c  = g >> 4;
            int ss = (g & 15) * 8;
#ifdef USE_ASYNC_V
            __builtin_amdgcn_global_load_async_to_lds_b128(
                (vsi4*)(vg + (long)c * SEQ + s0 + ss),
                (vsi4*)&vS[c * SCHUNK + ss], 0, 0);
#else
            uint4 v0 = *(const uint4*)(vg + (long)c * SEQ + s0 + ss);
            *(uint4*)&vS[c * SCHUNK + ss] = v0;
#endif
        }
#ifdef USE_ASYNC_V
        asm volatile("s_wait_asynccnt 0" ::: "memory");
#endif
        __syncthreads();

        // Prefetch next chunk (global_prefetch_b8): 64 rows x 2 lines each.
        int sn = s0 + SCHUNK;
        if (sn < SEQ && tid < 128) {
            int row = tid & 63, part = (tid >> 6) * 64;
            __builtin_prefetch(kg + (long)row * SEQ + sn + part, 0, 1);
            __builtin_prefetch(vg + (long)row * SEQ + sn + part, 0, 1);
        }

        // ---- S = Q^T K  (16t x 128s), A shared across the 8 N-tiles ----
        v8f sf[8];
#pragma unroll
        for (int j = 0; j < 8; ++j) sf[j] = zero8;
#pragma unroll
        for (int kk = 0; kk < 2; ++kk) {
            int k0 = kk * 32 + half * 8;
#pragma unroll
            for (int j = 0; j < 8; ++j) {
                const _Float16* bb = &kT[(j * 16 + lp) * CH];
                v16h bf = load_frag(bb + k0, bb + k0 + 16);
                sf[j] = __builtin_amdgcn_wmma_f32_16x16x32_f16(
                            false, qa[kk], false, bf, (short)0, sf[j],
                            false, false);
            }
        }

        // ---- online softmax (exp2 domain); max via permlane, sum via WMMA --
        float corr8[8];
#pragma unroll
        for (int r = 0; r < 8; ++r) {
            float cm = fmaxf(fmaxf(fmaxf(sf[0][r], sf[1][r]),
                                   fmaxf(sf[2][r], sf[3][r])),
                             fmaxf(fmaxf(sf[4][r], sf[5][r]),
                                   fmaxf(sf[6][r], sf[7][r])));
            cm = redmax16(cm);
            float mn = fmaxf(mrow[r], cm);
            corr8[r] = __builtin_amdgcn_exp2f(mrow[r] - mn);
            mrow[r] = mn;

            int m = r + 8 * half;
#pragma unroll
            for (int j = 0; j < 8; ++j) {
                float p = __builtin_amdgcn_exp2f(sf[j][r] - mn);
                pw[m * SCHUNK + j * 16 + lp] = (_Float16)p;
            }
#pragma unroll
            for (int j = 0; j < 4; ++j) o[j][r] *= corr8[r];
        }

        // ---- O += P V^T; l-chunk = P x ones on the WMMA pipe ----
        v8f lchunk = zero8;
#pragma unroll
        for (int kk = 0; kk < 4; ++kk) {             // K = s = 128 -> 4 steps
            int k0 = kk * 32 + half * 8;
            const _Float16* ab = &pw[lp * SCHUNK];
            v16h pa = load_frag(ab + k0, ab + k0 + 16);
            lchunk = __builtin_amdgcn_wmma_f32_16x16x32_f16(
                         false, pa, false, ones, (short)0, lchunk,
                         false, false);
#pragma unroll
            for (int j = 0; j < 4; ++j) {
                const _Float16* bb = &vS[(j * 16 + lp) * SCHUNK];
                v16h bf = load_frag(bb + k0, bb + k0 + 16);
                o[j] = __builtin_amdgcn_wmma_f32_16x16x32_f16(
                           false, pa, false, bf, (short)0, o[j],
                           false, false);
            }
        }
#pragma unroll
        for (int r = 0; r < 8; ++r)
            lrow[r] = lrow[r] * corr8[r] + lchunk[r];
    }

    // ---- epilogue: normalize, transpose through LDS, coalesced stores ----
    __syncthreads();                                 // staging LDS now dead
    const int tl = wid * 16;
#pragma unroll
    for (int r = 0; r < 8; ++r) {
        float inv = 1.0f / lrow[r];
        int t = tl + r + 8 * half;
#pragma unroll
        for (int j = 0; j < 4; ++j)
            obuf[(j * 16 + lp) * OSTRIDE + t] = o[j][r] * inv;
    }
    __syncthreads();
    for (int g = tid; g < 2048; g += 256) {          // 64 rows x 32 float4
        int c  = g >> 5;
        int tt = (g & 31) * 4;
        float4 val = *(float4*)&obuf[c * OSTRIDE + tt];
        *(float4*)&out[(long)b * (CH * SEQ) + (long)c * SEQ + t0 + tt] = val;
    }
}

// ---------------------------------------------------------------------------
extern "C" void kernel_launch(void* const* d_in, const int* in_sizes, int n_in,
                              void* d_out, int out_size, void* d_ws, size_t ws_size,
                              hipStream_t stream) {
    const float* qkv = (const float*)d_in[0];
    float* out = (float*)d_out;
    _Float16* qkvh = (_Float16*)d_ws;               // 50.3 MB f16 staging

    const long n_elems = 8L * 1536 * 2048;
    const long n4 = n_elems / 4;
    qkv_cvt_f16<<<(int)(n4 / 256), 256, 0, stream>>>(qkv, qkvh);

    dim3 grid(NHEADB, SEQ / TBLOCK);
    flash_attn_f16<<<grid, 256, 0, stream>>>(qkvh, out);
}
